// single_Mamba_31327491457240
// MI455X (gfx1250) — compile-verified, matrix operand order checked by reading, not run on hardware
//
#include <hip/hip_runtime.h>
#include <hip/hip_bf16.h>

// ---------------- problem constants (from reference) ----------------
enum : int {
    B_     = 8,
    L_     = 256,
    DM     = 256,      // D_MODEL
    DI     = 512,      // D_INNER
    DSTATE = 128,      // D_STATE
    DTR    = 16,       // DT_RANK
    NXD    = DTR + 2 * DSTATE,   // 272 columns of x_dbl
    BL     = B_ * L_,            // 2048 rows
};

typedef _Float16 v16h __attribute__((ext_vector_type(16)));
typedef float    v8f  __attribute__((ext_vector_type(8)));

union AFragU { v16h v; unsigned int u[8]; };
union CFragU { v8f  v; float f[8]; };

// ---------------- WMMA f16 GEMM: C(f32) = A(f16) @ Bpacked(f16) ----------------
// one wave32 per 16x16 C tile; K multiple of 32.
// B is pre-packed in fragment order: [ntile][kchunk][lane(32)][elem(16)] halves,
// so each lane's fragment is one contiguous 32-byte load.
template <bool CHECK_M>
__global__ __launch_bounds__(32)
void gemm_f16_wmma(const _Float16* __restrict__ A, const _Float16* __restrict__ Bp,
                   float* __restrict__ C, int M_valid, int K, int lda, int ldc)
{
    const int lane    = threadIdx.x & 31;
    const int tileN   = blockIdx.x;          // N/16 tiles
    const int tileM   = blockIdx.y;          // M/16 tiles
    const int r16     = lane & 15;
    const int g       = lane >> 4;
    const int kchunks = K >> 5;

    const _Float16* Arow  = A  + (tileM * 16 + r16) * lda;
    // base of this lane's packed B fragments for column-tile tileN
    const _Float16* Bfrag = Bp + ((size_t)tileN * kchunks * 32 + lane) * 16;

    v8f acc = {0.f, 0.f, 0.f, 0.f, 0.f, 0.f, 0.f, 0.f};

    #pragma unroll 2
    for (int c = 0; c < kchunks; ++c) {
        const int k0 = c * 32;
        __builtin_prefetch((const void*)(Arow + k0 + 32), 0, 1);  // global_prefetch_b8

        AFragU a;
        #pragma unroll
        for (int p = 0; p < 8; ++p) {
            // element pair (2p,2p+1): K = (p>=4)*16 + g*8 + (p&3)*2  (+k0)
            const int kk = k0 + ((p >> 2) * 16) + g * 8 + ((p & 3) * 2);
            a.u[p] = *(const unsigned int*)(Arow + kk);
        }
        const v16h bv = *(const v16h*)(Bfrag + (size_t)c * (32 * 16));

        acc = __builtin_amdgcn_wmma_f32_16x16x32_f16(
                  false, a.v, false, bv, (short)0, acc, false, false);
    }

    CFragU cf; cf.v = acc;
    float* Cb = C + (tileM * 16 + g * 8) * ldc + tileN * 16 + r16;
    #pragma unroll
    for (int r = 0; r < 8; ++r) {
        if (!CHECK_M || (tileM * 16 + g * 8 + r) < M_valid)
            Cb[r * ldc] = cf.f[r];
    }
}

// ---------------- pack f32 weight matrix into WMMA B-fragment order ----------------
// dst layout: [ntile = N/16][kchunk = Kpad/32][lane 0..31][elem 0..15] f16.
// Rows k >= srcRows read as 0 (used to pad W_dt K 16 -> 32).
__global__ void pack_bfrag(const float* __restrict__ src, _Float16* __restrict__ dst,
                           int Kpad, int srcRows, int srcLd)
{
    const int i = blockIdx.x * blockDim.x + threadIdx.x;
    const int kchunks = Kpad >> 5;
    // total elements = N * Kpad (launched exactly)
    const int e     = i & 15;
    const int lane  = (i >> 4) & 31;
    const int rest  = i >> 9;
    const int chunk = rest % kchunks;
    const int ntile = rest / kchunks;

    const int k   = chunk * 32 + ((e >> 3) * 16) + ((lane >> 4) * 8) + (e & 7);
    const int col = ntile * 16 + (lane & 15);
    const float v = (k < srcRows) ? src[(size_t)k * srcLd + col] : 0.f;
    dst[i] = (_Float16)v;
}

// ---------------- embedding gather + time term -> f16 ----------------
__global__ void embed_kernel(const int* __restrict__ poi, const int* __restrict__ cat,
                             const int* __restrict__ hour, const float* __restrict__ tsec,
                             const float* __restrict__ poi_emb, const float* __restrict__ cat_emb,
                             const float* __restrict__ hour_emb, const float* __restrict__ time_w,
                             _Float16* __restrict__ hs_h)
{
    const int i = blockIdx.x * blockDim.x + threadIdx.x;
    if (i >= BL * DM) return;
    const int r = i >> 8;
    const int c = i & 255;
    const float v = poi_emb[(size_t)poi[r]  * DM + c]
                  + cat_emb[(size_t)cat[r]  * DM + c]
                  + hour_emb[(size_t)hour[r] * DM + c]
                  + tsec[r] * time_w[c];
    hs_h[i] = (_Float16)v;
}

// ---------------- f32 -> f16 pack with row/col zero padding (A matrices) ----------------
__global__ void pack_f16_pad(const float* __restrict__ src, _Float16* __restrict__ dst,
                             int srcRows, int srcCols, int srcLd,
                             int dstRows, int dstCols)
{
    const int i = blockIdx.x * blockDim.x + threadIdx.x;
    if (i >= dstRows * dstCols) return;
    const int r = i / dstCols, c = i % dstCols;
    const float v = (r < srcRows && c < srcCols) ? src[(size_t)r * srcLd + c] : 0.f;
    dst[i] = (_Float16)v;
}

// ---------------- depthwise conv (D_CONV=2) + SiLU ----------------
__global__ void conv_silu_kernel(const float* __restrict__ xz, const float* __restrict__ conv_w,
                                 const float* __restrict__ conv_b,
                                 float* __restrict__ xc, _Float16* __restrict__ xc_h)
{
    const int i = blockIdx.x * blockDim.x + threadIdx.x;
    if (i >= BL * DI) return;
    const int r = i / DI, d = i % DI;
    const int l = r & (L_ - 1);
    const float xcur  = xz[(size_t)r * (2 * DI) + d];
    const float xprev = (l > 0) ? xz[(size_t)(r - 1) * (2 * DI) + d] : 0.f;
    float v = conv_b[d] + conv_w[d * 2 + 0] * xprev + conv_w[d * 2 + 1] * xcur;
    v = v / (1.f + __expf(-v));            // silu
    xc[i]   = v;
    xc_h[i] = (_Float16)v;
}

// ---------------- selective scan, sharded along d ----------------
// grid = (B, DI/32); block = 256 threads; thread owns (1 d, 16 n) of state.
__global__ __launch_bounds__(256)
void scan_kernel(const float* __restrict__ dtpre, const float* __restrict__ b_dt,
                 const float* __restrict__ xc, const float* __restrict__ xdbl,
                 const float* __restrict__ A_log, float* __restrict__ yscan)
{
    __shared__ float dt_s[32], xc_s[32], B_s[DSTATE];
    const int b  = blockIdx.x;
    const int d0 = blockIdx.y * 32;
    const int t  = threadIdx.x;
    const int dl = t >> 3, k = t & 7;
    const int d  = d0 + dl, n0 = k * 16;

    float Arow[16], h[16];
    #pragma unroll
    for (int j = 0; j < 16; ++j) {
        Arow[j] = -__expf(A_log[d * DSTATE + n0 + j]);
        h[j] = 0.f;
    }

    for (int l = 0; l < L_; ++l) {
        const size_t row = (size_t)(b * L_ + l);
        __syncthreads();
        if (t < 32) {
            const float p = dtpre[row * DI + d0 + t] + b_dt[d0 + t];
            dt_s[t] = (p > 20.f) ? p : __logf(1.f + __expf(p));   // softplus
        } else if (t < 64) {
            xc_s[t - 32] = xc[row * DI + d0 + (t - 32)];
        } else if (t < 64 + DSTATE) {
            B_s[t - 64] = xdbl[row * NXD + DTR + (t - 64)];
        }
        __syncthreads();
        const float dtv = dt_s[dl];
        const float cm  = dtv * xc_s[dl];
        #pragma unroll
        for (int j = 0; j < 16; ++j)
            h[j] = __expf(dtv * Arow[j]) * h[j] + cm * B_s[n0 + j];
    }

    // y_d = sum_n h[d,n] * C[last, n]  (reduce across the 8 lanes sharing d)
    const float* Crow = xdbl + (size_t)(b * L_ + (L_ - 1)) * NXD + DTR + DSTATE;
    float part = 0.f;
    #pragma unroll
    for (int j = 0; j < 16; ++j) part += h[j] * Crow[n0 + j];
    part += __shfl_down(part, 4, 8);
    part += __shfl_down(part, 2, 8);
    part += __shfl_down(part, 1, 8);
    if (k == 0) yscan[b * DI + d] = part;
}

// ---------------- final gating at last step -> f16 A-matrix (16x512) ----------------
__global__ void yfin_kernel(const float* __restrict__ yscan, const float* __restrict__ D_param,
                            const float* __restrict__ xc, const float* __restrict__ xz,
                            _Float16* __restrict__ yfin_h)
{
    const int i = blockIdx.x * blockDim.x + threadIdx.x;
    if (i >= 16 * DI) return;
    const int r = i / DI, d = i % DI;
    float v = 0.f;
    if (r < B_) {
        const size_t last = (size_t)(r * L_ + (L_ - 1));
        const float y = yscan[r * DI + d] + D_param[d] * xc[last * DI + d];
        const float z = xz[last * (2 * DI) + DI + d];
        v = y * (z / (1.f + __expf(-z)));   // y * silu(z)
    }
    yfin_h[i] = (_Float16)v;
}

// ---------------- host side ----------------
extern "C" void kernel_launch(void* const* d_in, const int* in_sizes, int n_in,
                              void* d_out, int out_size, void* d_ws, size_t ws_size,
                              hipStream_t stream)
{
    (void)in_sizes; (void)n_in; (void)out_size; (void)ws_size;

    const int*   poi      = (const int*)  d_in[0];
    const int*   cat      = (const int*)  d_in[1];
    const int*   hour     = (const int*)  d_in[2];
    const float* tsec     = (const float*)d_in[3];
    /* d_in[4] = len_seq, unused by reference */
    const float* poi_emb  = (const float*)d_in[5];
    const float* cat_emb  = (const float*)d_in[6];
    const float* hour_emb = (const float*)d_in[7];
    const float* time_w   = (const float*)d_in[8];
    const float* W_in     = (const float*)d_in[9];
    const float* conv_w   = (const float*)d_in[10];
    const float* conv_b   = (const float*)d_in[11];
    const float* W_x      = (const float*)d_in[12];
    const float* W_dt     = (const float*)d_in[13];
    const float* b_dt     = (const float*)d_in[14];
    const float* A_log    = (const float*)d_in[15];
    const float* D_param  = (const float*)d_in[16];
    const float* W_out    = (const float*)d_in[17];
    float* out            = (float*)d_out;

    char* ws = (char*)d_ws;
    size_t off = 0;
    auto suballoc = [&](size_t bytes) -> void* {
        void* p = ws + off;
        off += (bytes + 255) & ~(size_t)255;
        return p;
    };

    _Float16* hs_h    = (_Float16*)suballoc((size_t)BL * DM * 2);
    _Float16* Win_bp  = (_Float16*)suballoc((size_t)DM * 2 * DI * 2);   // 256 x 1024 packed
    _Float16* Wx_bp   = (_Float16*)suballoc((size_t)DI * NXD * 2);      // 512 x 272 packed
    _Float16* Wdt_bp  = (_Float16*)suballoc((size_t)32 * DI * 2);       // 32(pad) x 512 packed
    _Float16* Wout_bp = (_Float16*)suballoc((size_t)DI * DM * 2);       // 512 x 256 packed
    float*    xz      = (float*)   suballoc((size_t)BL * 2 * DI * 4);
    float*    xc      = (float*)   suballoc((size_t)BL * DI * 4);
    _Float16* xc_h    = (_Float16*)suballoc((size_t)BL * DI * 2);
    float*    xdbl    = (float*)   suballoc((size_t)BL * NXD * 4);
    _Float16* dtlow_h = (_Float16*)suballoc((size_t)BL * 32 * 2);       // A cols padded 16->32
    float*    dtpre   = (float*)   suballoc((size_t)BL * DI * 4);
    float*    yscan   = (float*)   suballoc((size_t)B_ * DI * 4);
    _Float16* yfin_h  = (_Float16*)suballoc((size_t)16 * DI * 2);

    const int TB = 256;

    // 1) embeddings -> hs (f16)
    embed_kernel<<<(BL * DM + TB - 1) / TB, TB, 0, stream>>>(
        poi, cat, hour, tsec, poi_emb, cat_emb, hour_emb, time_w, hs_h);

    // 2) pack weights into WMMA B-fragment order (f16, padded where needed)
    pack_bfrag<<<(DM * 2 * DI) / TB, TB, 0, stream>>>(W_in,  Win_bp,  DM, DM,  2 * DI);
    pack_bfrag<<<(DI * NXD)   / TB, TB, 0, stream>>>(W_x,   Wx_bp,   DI, DI,  NXD);
    pack_bfrag<<<(32 * DI)    / TB, TB, 0, stream>>>(W_dt,  Wdt_bp,  32, DTR, DI);
    pack_bfrag<<<(DI * DM)    / TB, TB, 0, stream>>>(W_out, Wout_bp, DI, DI,  DM);

    // 3) xz = hs @ W_in   (2048 x 1024, K=256)
    gemm_f16_wmma<false><<<dim3((2 * DI) / 16, BL / 16), 32, 0, stream>>>(
        hs_h, Win_bp, xz, BL, DM, DM, 2 * DI);

    // 4) depthwise conv + silu -> xc (f32 + f16)
    conv_silu_kernel<<<(BL * DI + TB - 1) / TB, TB, 0, stream>>>(xz, conv_w, conv_b, xc, xc_h);

    // 5) x_dbl = xc @ W_x  (2048 x 272, K=512)
    gemm_f16_wmma<false><<<dim3(NXD / 16, BL / 16), 32, 0, stream>>>(
        xc_h, Wx_bp, xdbl, BL, DI, DI, NXD);

    // 6) dt_low (first 16 cols of x_dbl) -> f16 A matrix, K padded to 32
    pack_f16_pad<<<(BL * 32 + TB - 1) / TB, TB, 0, stream>>>(xdbl, dtlow_h, BL, DTR, NXD, BL, 32);

    // 7) dt_pre = dt_low @ W_dt  (2048 x 512, K=32 incl. zero pad)
    gemm_f16_wmma<false><<<dim3(DI / 16, BL / 16), 32, 0, stream>>>(
        dtlow_h, Wdt_bp, dtpre, BL, 32, 32, DI);

    // 8) selective scan (softplus fused); only final-step y is produced
    scan_kernel<<<dim3(B_, DI / 32), 256, 0, stream>>>(dtpre, b_dt, xc, xdbl, A_log, yscan);

    // 9) gate with silu(z_last), add D*xc_last -> f16 (rows 8..15 zero)
    yfin_kernel<<<(16 * DI + TB - 1) / TB, TB, 0, stream>>>(yscan, D_param, xc, xz, yfin_h);

    // 10) out = yfin @ W_out  (8 x 256, K=512) written straight to d_out
    gemm_f16_wmma<true><<<dim3(DM / 16, 1), 32, 0, stream>>>(
        yfin_h, Wout_bp, out, B_, DI, DI, DM);
}
